// HGCN_21723944583573
// MI455X (gfx1250) — matrix operand order repaired and checked by reading, not compiled, not run on hardware
//
#include <hip/hip_runtime.h>
#include <cstdint>

// ---------------------------------------------------------------------------
// Problem dims (fixed by the reference)
// ---------------------------------------------------------------------------
constexpr int NDIM = 16, CDIM = 32, VDIM = 2048, VCDIM = 256, SDIM = 64, LDIM = 12, CODIM = 32;
constexpr int FDIM = NDIM * CDIM * LDIM;          // 6144  (== NDIM*CODIM*LDIM)
constexpr float N1S = 0.8f, N2S = 0.2f, N3S = 0.2f, N4S = 0.2f;

typedef __attribute__((ext_vector_type(16))) __bf16 v16bf;
typedef __attribute__((ext_vector_type(8)))  float  v8f;

#define DEVFN __device__ __forceinline__

DEVFN __bf16 f2bf(float x) {
    unsigned u = __builtin_bit_cast(unsigned, x);
    u += 0x7FFFu + ((u >> 16) & 1u);               // round-to-nearest-even
    unsigned short h = (unsigned short)(u >> 16);
    return __builtin_bit_cast(__bf16, h);
}
DEVFN float bf2f(__bf16 b) {
    unsigned short h = __builtin_bit_cast(unsigned short, b);
    unsigned u = ((unsigned)h) << 16;
    return __builtin_bit_cast(float, u);
}

// CDNA5 async global->LDS copy (ASYNCcnt path, ISA 15.18.3 op 98).
// dst: LDS byte offset (per-lane VGPR); src: 64-bit global address.
DEVFN void async_g2lds_b128(unsigned ldsOff, const void* src) {
    asm volatile("global_load_async_to_lds_b128 %0, %1, off"
                 :: "v"(ldsOff), "v"((unsigned long long)(uintptr_t)src)
                 : "memory");
}
DEVFN void wait_asynccnt0() {
#if __has_builtin(__builtin_amdgcn_s_wait_asynccnt)
    __builtin_amdgcn_s_wait_asynccnt(0);
#else
    asm volatile("s_wait_asynccnt 0x0" ::: "memory");
#endif
}

// ---------------------------------------------------------------------------
// Generic bf16 WMMA GEMM:  C[m,f] = sum_k A(m,k) * B[f][k]   (+ optional 2nd
// A2/B2 pair accumulated into the same tile). A is fp32 (opt. transposed),
// converted to bf16 on the fly. B is bf16 in K-minor layout [FDIM][K]
// (row stride K) so a 128x32 tile is 512 contiguous 16B chunks -> staged
// with GLOBAL_LOAD_ASYNC_TO_LDS_B128.
//
// Epilogues (template EPI):
//  0: outB[f*M+m]  = bf16(acc)                                  (xcT, K-minor)
//  1: outF[m*F+f] = acc ; outB[f*M+m] = bf16(acc)               (sxg)
//  2: outF[m*F+f]  = acc + bf2f(U[f*M+m]) + bias[o]             (hc/hs pre)
//  3: d_out(hf map)= acc + bf2f(U[f*M+m]) + bias[o]             (hf pre)
//  4: outF[m*F+f] += scale * relu(acc)                          (ws fusion)
//  5: d_out(hf map)+= scale * relu(acc)                         (hf fusion)
// ---------------------------------------------------------------------------
constexpr int BM = 128, BFT = 128, BK = 32, LPAD = 8;   // As row = 40 bf16 = 80B (16B-aligned)

template <int EPI>
__global__ __launch_bounds__(256)
void gemm_wmma(int M, int K,
               const float* A1, int lda1, int tA1, const __bf16* B1,
               const float* A2, int lda2, int tA2, const __bf16* B2,
               const __bf16* Uadd, const float* bias,
               float* outF, __bf16* outB, float scale)
{
    __shared__ __attribute__((aligned(16))) __bf16 As[BM][BK + LPAD];
    __shared__ __attribute__((aligned(16))) __bf16 Bs[BFT][BK];     // [f][k], rows 64B

    const int tid  = threadIdx.x;
    const int lane = tid & 31;
    const int wave = tid >> 5;
    const int wm   = wave & 3;        // 4 waves tile M (32 rows each)
    const int wf   = wave >> 2;       // 2 waves tile F (64 cols each)
    const int m0   = blockIdx.y * BM;
    const int f0   = blockIdx.x * BFT;
    const int lm   = lane & 15;
    const int hi   = lane >> 4;       // 0 or 1

    const unsigned bsBase = (unsigned)(uintptr_t)&Bs[0][0];  // low 32 bits = LDS offset

    v8f acc[2][4];
    const v8f vz = {0.f,0.f,0.f,0.f,0.f,0.f,0.f,0.f};
#pragma unroll
    for (int i = 0; i < 2; ++i)
#pragma unroll
        for (int j = 0; j < 4; ++j) acc[i][j] = vz;

    for (int ph = 0; ph < 2; ++ph) {
        const float*  A     = ph ? A2 : A1;
        const __bf16* B     = ph ? B2 : B1;
        const int     trans = ph ? tA2 : tA1;
        const int     lda   = ph ? lda2 : lda1;
        if (A == nullptr) continue;

        for (int k0 = 0; k0 < K; k0 += BK) {
            // ---- stage B tile via async copy unit: 512 x 16B chunks ----
#pragma unroll
            for (int p = 0; p < 2; ++p) {
                const int chunk = tid + p * 256;
                const int f  = chunk >> 2;          // 0..127
                const int c8 = (chunk & 3) << 3;    // k offset: 0,8,16,24
                const __bf16* src = B + (size_t)(f0 + f) * K + (k0 + c8);
                const unsigned dst = bsBase + (unsigned)((f * BK + c8) * 2);
                async_g2lds_b128(dst, src);
            }
            // ---- stage A tile (fp32 -> bf16, optional transpose) ----
            for (int idx = tid; idx < BM * BK; idx += 256) {
                int m, k;
                if (trans) { m = idx & (BM - 1); k = idx / BM; }   // coalesced over m
                else       { k = idx & (BK - 1); m = idx / BK; }   // coalesced over k
                const int gm = m0 + m;
                float v = 0.f;
                if (gm < M)
                    v = trans ? A[(size_t)(k0 + k) * lda + gm]
                              : A[(size_t)gm * lda + (k0 + k)];
                As[m][k] = f2bf(v);
            }
            wait_asynccnt0();
            __syncthreads();

            union Frg { uint4 q[2]; v16bf v; };
            Frg fa[2], fb[4];
            // A frag: lanes 0-15 -> K = 0..7 / 16..23 ; lanes 16-31 -> K = 8..15 / 24..31
#pragma unroll
            for (int tm = 0; tm < 2; ++tm) {
                const __bf16* p = &As[wm * 32 + tm * 16 + lm][hi * 8];
                fa[tm].q[0] = *(const uint4*)(p);
                fa[tm].q[1] = *(const uint4*)(p + 16);
            }
            // B frag: lanes 0-15 -> K = 0..7 / 16..23 ; lanes 16-31 -> K = 8..15 / 24..31
#pragma unroll
            for (int tf = 0; tf < 4; ++tf) {
                const __bf16* p = &Bs[wf * 64 + tf * 16 + lm][hi * 8];
                fb[tf].q[0] = *(const uint4*)(p);
                fb[tf].q[1] = *(const uint4*)(p + 16);
            }
#pragma unroll
            for (int tm = 0; tm < 2; ++tm)
#pragma unroll
                for (int tf = 0; tf < 4; ++tf)
                    acc[tm][tf] = __builtin_amdgcn_wmma_f32_16x16x32_bf16(
                        false, fa[tm].v, false, fb[tf].v,
                        (short)0, acc[tm][tf], false, false);
            __syncthreads();
        }
    }

    // ---- epilogue (C layout: VGPR r -> M = r + 8*hi, N = lane&15) ----
#pragma unroll
    for (int tm = 0; tm < 2; ++tm) {
        const int mbase = m0 + wm * 32 + tm * 16 + hi * 8;
#pragma unroll
        for (int tf = 0; tf < 4; ++tf) {
            const int f = f0 + wf * 64 + tf * 16 + lm;
#pragma unroll
            for (int r = 0; r < 8; ++r) {
                const int m = mbase + r;
                if (m >= M) continue;
                float v = acc[tm][tf][r];
                const size_t mf = (size_t)m * FDIM + f;
                const size_t fm = (size_t)f * M + m;     // K-minor index
                if (EPI == 0) {
                    outB[fm] = f2bf(v);
                } else if (EPI == 1) {
                    outF[mf] = v; outB[fm] = f2bf(v);
                } else if (EPI == 2) {
                    const int o = (f / LDIM) % CODIM;
                    outF[mf] = v + bf2f(Uadd[fm]) + bias[o];
                } else if (EPI == 3) {
                    const int n = f / (CODIM * LDIM);
                    const int o = (f / LDIM) % CODIM;
                    const int l = f % LDIM;
                    outF[(((size_t)n * CODIM + o) * M + m) * LDIM + l] =
                        v + bf2f(Uadd[fm]) + bias[o];
                } else if (EPI == 4) {
                    outF[mf] += scale * fmaxf(v, 0.f);
                } else if (EPI == 5) {
                    const int n = f / (CODIM * LDIM);
                    const int o = (f / LDIM) % CODIM;
                    const int l = f % LDIM;
                    const size_t oi = (((size_t)n * CODIM + o) * M + m) * LDIM + l;
                    outF[oi] += scale * fmaxf(v, 0.f);
                }
            }
        }
    }
}

// ---------------------------------------------------------------------------
// Small helper kernels (all "B-side" bf16 tensors K-minor: [FDIM][nodes])
// ---------------------------------------------------------------------------
// x [N,C,V,L] fp32 -> XtT bf16 [F][V], f = (n*C+c)*L + l
__global__ void pack_x_kernel(const float* __restrict__ x, __bf16* __restrict__ XtT)
{
    const size_t idx = (size_t)blockIdx.x * blockDim.x + threadIdx.x;
    if (idx >= (size_t)VDIM * FDIM) return;
    const int fo = (int)(idx / VDIM);
    const int v  = (int)(idx % VDIM);
    const int n = fo / (CDIM * LDIM), c = (fo / LDIM) % CDIM, l = fo % LDIM;
    XtT[idx] = f2bf(x[(((size_t)n * CDIM + c) * VDIM + v) * LDIM + l]);
}

// UjT[f(n,o,l)][v] = sum_c W[o][j*32+c] * XT[f(n,c,l)][v]   for j=0,1,2
__global__ void channel_mix_kernel(const __bf16* __restrict__ XT, const float* __restrict__ W,
                                   __bf16* __restrict__ U1, __bf16* __restrict__ U2,
                                   __bf16* __restrict__ U3, int nodes)
{
    const size_t idx = (size_t)blockIdx.x * blockDim.x + threadIdx.x;
    if (idx >= (size_t)nodes * FDIM) return;
    const int fo = (int)(idx / nodes);
    const int v  = (int)(idx % nodes);
    const int n = fo / (CODIM * LDIM), o = (fo / LDIM) % CODIM, l = fo % LDIM;
    const float* wr = W + (size_t)o * (3 * CDIM);
    const size_t fbase = (size_t)n * CDIM * LDIM + l;
    float a1 = 0.f, a2 = 0.f, a3 = 0.f;
#pragma unroll 8
    for (int c = 0; c < CDIM; ++c) {
        const float xv = bf2f(XT[(fbase + (size_t)c * LDIM) * nodes + v]);
        a1 += wr[c] * xv; a2 += wr[CDIM + c] * xv; a3 += wr[2 * CDIM + c] * xv;
    }
    U1[idx] = f2bf(a1); U2[idx] = f2bf(a2); U3[idx] = f2bf(a3);
}

// as_mat[s1][s2] = relu( sum_k sxg[s1][f1(k)] * sxg[s2][f2(k)] - 0.5 )
// k decoded per the mismatched torch flatten orders in the reference.
__global__ void gram_kernel(const float* __restrict__ sxg, float* __restrict__ am)
{
    const int s1 = blockIdx.x / SDIM, s2 = blockIdx.x % SDIM;
    __shared__ float red[128];
    float p = 0.f;
    for (int k = threadIdx.x; k < FDIM; k += 128) {
        const int l1 = k % LDIM, r1 = k / LDIM;
        const int n1 = r1 % NDIM, c1 = r1 / NDIM;
        const int f1 = (n1 * CDIM + c1) * LDIM + l1;
        const int c2 = k % CDIM, r2 = k / CDIM;
        const int n2 = r2 % NDIM, l2 = r2 / NDIM;
        const int f2 = (n2 * CDIM + c2) * LDIM + l2;
        p += sxg[(size_t)s1 * FDIM + f1] * sxg[(size_t)s2 * FDIM + f2];
    }
    red[threadIdx.x] = p; __syncthreads();
    for (int s = 64; s > 0; s >>= 1) {
        if ((int)threadIdx.x < s) red[threadIdx.x] += red[threadIdx.x + s];
        __syncthreads();
    }
    if (threadIdx.x == 0) am[s1 * SDIM + s2] = fmaxf(red[0] - 0.5f, 0.f);
}

// sup = softmax(asym_adj(as_mat)) rows (mode 0) / of as_mat^T (mode 1)
__global__ void make_sup_kernel(const float* __restrict__ am,
                                float* __restrict__ sup0, float* __restrict__ sup1)
{
    const int s = blockIdx.x, mode = blockIdx.y, t = threadIdx.x;
    __shared__ float red[64];
    const float v = mode ? am[t * SDIM + s] : am[s * SDIM + t];
    red[t] = v; __syncthreads();
    for (int w = 32; w > 0; w >>= 1) { if (t < w) red[t] += red[t + w]; __syncthreads(); }
    const float rowsum = red[0]; __syncthreads();
    const float dinv = rowsum > 0.f ? 1.f / rowsum : 0.f;
    const float a = v * dinv;
    red[t] = a; __syncthreads();
    for (int w = 32; w > 0; w >>= 1) { if (t < w) red[t] = fmaxf(red[t], red[t + w]); __syncthreads(); }
    const float mx = red[0]; __syncthreads();
    const float e = __expf(a - mx);
    red[t] = e; __syncthreads();
    for (int w = 32; w > 0; w >>= 1) { if (t < w) red[t] += red[t + w]; __syncthreads(); }
    (mode ? sup1 : sup0)[s * SDIM + t] = e / red[0];
}

// ws [nodes][F] fp32 -> outT [F][nodes] bf16
__global__ void cvt_f32_bf16T_kernel(const float* __restrict__ in, __bf16* __restrict__ outT,
                                     int nodes)
{
    const size_t idx = (size_t)blockIdx.x * blockDim.x + threadIdx.x;
    if (idx >= (size_t)nodes * FDIM) return;
    const int fo = (int)(idx / nodes);
    const int m  = (int)(idx % nodes);
    outT[idx] = f2bf(in[(size_t)m * FDIM + fo]);
}

// d_out hf [N,CO,V,L] fp32 -> hf_bf [F][V] bf16
__global__ void hf_to_bf_kernel(const float* __restrict__ hf, __bf16* __restrict__ outT)
{
    const size_t idx = (size_t)blockIdx.x * blockDim.x + threadIdx.x;
    if (idx >= (size_t)VDIM * FDIM) return;
    const int fo = (int)(idx / VDIM);
    const int v  = (int)(idx % VDIM);
    const int n = fo / (CODIM * LDIM), o = (fo / LDIM) % CODIM, l = fo % LDIM;
    outT[idx] = f2bf(hf[(((size_t)n * CODIM + o) * VDIM + v) * LDIM + l]);
}

// ws [nodes][F] fp32 -> d_out region [N,CO,nodes,L]
__global__ void unpack_kernel(const float* __restrict__ ws, float* __restrict__ out, int nodes)
{
    const size_t idx = (size_t)blockIdx.x * blockDim.x + threadIdx.x;
    if (idx >= (size_t)nodes * FDIM) return;
    const int w = (int)(idx / FDIM);
    const int fo = (int)(idx % FDIM);
    const int n = fo / (CODIM * LDIM), o = (fo / LDIM) % CODIM, l = fo % LDIM;
    out[(((size_t)n * CODIM + o) * nodes + w) * LDIM + l] = ws[idx];
}

// ---------------------------------------------------------------------------
// Host side
// ---------------------------------------------------------------------------
static inline size_t cdiv(size_t a, size_t b) { return (a + b - 1) / b; }

static void launch_gemm(int epi, int M, int K,
                        const float* A1, int lda1, int tA1, const __bf16* B1,
                        const float* A2, int lda2, int tA2, const __bf16* B2,
                        const __bf16* U, const float* bias,
                        float* outF, __bf16* outB, float scale, hipStream_t s)
{
    dim3 grid(FDIM / BFT, (unsigned)cdiv(M, BM));
    switch (epi) {
    case 0: gemm_wmma<0><<<grid, 256, 0, s>>>(M, K, A1, lda1, tA1, B1, A2, lda2, tA2, B2, U, bias, outF, outB, scale); break;
    case 1: gemm_wmma<1><<<grid, 256, 0, s>>>(M, K, A1, lda1, tA1, B1, A2, lda2, tA2, B2, U, bias, outF, outB, scale); break;
    case 2: gemm_wmma<2><<<grid, 256, 0, s>>>(M, K, A1, lda1, tA1, B1, A2, lda2, tA2, B2, U, bias, outF, outB, scale); break;
    case 3: gemm_wmma<3><<<grid, 256, 0, s>>>(M, K, A1, lda1, tA1, B1, A2, lda2, tA2, B2, U, bias, outF, outB, scale); break;
    case 4: gemm_wmma<4><<<grid, 256, 0, s>>>(M, K, A1, lda1, tA1, B1, A2, lda2, tA2, B2, U, bias, outF, outB, scale); break;
    case 5: gemm_wmma<5><<<grid, 256, 0, s>>>(M, K, A1, lda1, tA1, B1, A2, lda2, tA2, B2, U, bias, outF, outB, scale); break;
    }
}

extern "C" void kernel_launch(void* const* d_in, const int* in_sizes, int n_in,
                              void* d_out, int out_size, void* d_ws, size_t ws_size,
                              hipStream_t stream)
{
    (void)in_sizes; (void)n_in; (void)out_size; (void)ws_size;

    const float* x         = (const float*)d_in[0];   // [N,C,V,L]
    const float* support   = (const float*)d_in[1];   // [2,V,V]
    const float* support_c = (const float*)d_in[2];   // [2,VC,VC]
    const float* acs       = (const float*)d_in[3];   // [VC,S]
    const float* afc       = (const float*)d_in[4];   // [V,VC]
    const float* W         = (const float*)d_in[5];   // [CO,3C]
    const float* bias      = (const float*)d_in[6];   // [CO]

    float* out    = (float*)d_out;
    float* hf_out = out;
    float* hc_out = hf_out + (size_t)NDIM * CODIM * VDIM * LDIM;
    float* hs_out = hc_out + (size_t)NDIM * CODIM * VCDIM * LDIM;

    const size_t EL_VF  = (size_t)VDIM  * FDIM;   // 12,582,912
    const size_t EL_VCF = (size_t)VCDIM * FDIM;   //  1,572,864
    const size_t EL_SF  = (size_t)SDIM  * FDIM;   //    393,216

    char* wsp = (char*)d_ws;
    auto alloc = [&](size_t bytes) -> char* {
        char* p = wsp; wsp += (bytes + 255) & ~(size_t)255; return p;
    };
    __bf16* Xt    = (__bf16*)alloc(EL_VF  * 2);   // [F][V]
    __bf16* U1    = (__bf16*)alloc(EL_VF  * 2);   // [F][V]
    __bf16* U2    = (__bf16*)alloc(EL_VF  * 2);   // reused as hf_bf after hf-pre
    __bf16* U3    = (__bf16*)alloc(EL_VF  * 2);
    __bf16* xcT   = (__bf16*)alloc(EL_VCF * 2);   // [F][VC]
    __bf16* U1c   = (__bf16*)alloc(EL_VCF * 2);
    __bf16* U2c   = (__bf16*)alloc(EL_VCF * 2);
    __bf16* U3c   = (__bf16*)alloc(EL_VCF * 2);
    float*  sxgF  = (float*) alloc(EL_SF  * 4);   // [S][F]
    __bf16* sxgB  = (__bf16*)alloc(EL_SF  * 2);   // [F][S]
    __bf16* U1s   = (__bf16*)alloc(EL_SF  * 2);
    __bf16* U2s   = (__bf16*)alloc(EL_SF  * 2);
    __bf16* U3s   = (__bf16*)alloc(EL_SF  * 2);
    float*  amat  = (float*) alloc((size_t)SDIM * SDIM * 4);
    float*  sup0  = (float*) alloc((size_t)SDIM * SDIM * 4);
    float*  sup1  = (float*) alloc((size_t)SDIM * SDIM * 4);
    float*  hc_ws = (float*) alloc(EL_VCF * 4);   // [VC][F]
    float*  hs_ws = (float*) alloc(EL_SF  * 4);   // [S][F]
    __bf16* hc_bf = (__bf16*)alloc(EL_VCF * 2);   // [F][VC]
    __bf16* hs_bf = (__bf16*)alloc(EL_SF  * 2);   // [F][S]
    __bf16* hf_bf = U2;                            // alias: U2/U3 dead after hf-pre

    // 1) pack x -> XtT [F][V] bf16
    pack_x_kernel<<<cdiv(EL_VF, 256), 256, 0, stream>>>(x, Xt);
    // 2) channel mix fine: U1/U2/U3 = W1x / W2x / W3x   ([F][V])
    channel_mix_kernel<<<cdiv(EL_VF, 256), 256, 0, stream>>>(Xt, W, U1, U2, U3, VDIM);
    // 3) xc = afc^T x        (M=VC, K=V)  -> xcT [F][VC]
    launch_gemm(0, VCDIM, VDIM, afc, VCDIM, 1, Xt,
                nullptr, 0, 0, nullptr, nullptr, nullptr, nullptr, xcT, 0.f, stream);
    // 4) channel mix coarse  ([F][VC])
    channel_mix_kernel<<<cdiv(EL_VCF, 256), 256, 0, stream>>>(xcT, W, U1c, U2c, U3c, VCDIM);
    // 5) sxg = acs^T xc      (M=S, K=VC) -> fp32 [S][F] + bf16 [F][S]
    launch_gemm(1, SDIM, VCDIM, acs, SDIM, 1, xcT,
                nullptr, 0, 0, nullptr, nullptr, nullptr, sxgF, sxgB, 0.f, stream);
    // 6) channel mix super   ([F][S])
    channel_mix_kernel<<<cdiv(EL_SF, 256), 256, 0, stream>>>(sxgB, W, U1s, U2s, U3s, SDIM);
    // 7) data-dependent super adjacency
    gram_kernel<<<SDIM * SDIM, 128, 0, stream>>>(sxgF, amat);
    make_sup_kernel<<<dim3(SDIM, 2), 64, 0, stream>>>(amat, sup0, sup1);
    // 8) hf_pre = A1^T U2 + A2^T U3 + U1 + b  -> d_out (mapped)
    launch_gemm(3, VDIM, VDIM, support, VDIM, 1, U2,
                support + (size_t)VDIM * VDIM, VDIM, 1, U3,
                U1, bias, hf_out, nullptr, 0.f, stream);
    // 9) hc_pre = Ac1^T U2c + Ac2^T U3c + U1c + b -> hc_ws
    launch_gemm(2, VCDIM, VCDIM, support_c, VCDIM, 1, U2c,
                support_c + (size_t)VCDIM * VCDIM, VCDIM, 1, U3c,
                U1c, bias, hc_ws, nullptr, 0.f, stream);
    // 10) hs_pre = As1^T U2s + As2^T U3s + U1s + b -> hs_ws
    launch_gemm(2, SDIM, SDIM, sup0, SDIM, 1, U2s,
                sup1, SDIM, 1, U3s, U1s, bias, hs_ws, nullptr, 0.f, stream);
    // 11) hc += N1 * relu(acs · hs)
    cvt_f32_bf16T_kernel<<<cdiv(EL_SF, 256), 256, 0, stream>>>(hs_ws, hs_bf, SDIM);
    launch_gemm(4, VCDIM, SDIM, acs, SDIM, 0, hs_bf,
                nullptr, 0, 0, nullptr, nullptr, nullptr, hc_ws, nullptr, N1S, stream);
    // 12) hf += N2 * relu(afc · hc)
    cvt_f32_bf16T_kernel<<<cdiv(EL_VCF, 256), 256, 0, stream>>>(hc_ws, hc_bf, VCDIM);
    launch_gemm(5, VDIM, VCDIM, afc, VCDIM, 0, hc_bf,
                nullptr, 0, 0, nullptr, nullptr, nullptr, hf_out, nullptr, N2S, stream);
    // 13) hc += N3 * relu(afc^T · hf)
    hf_to_bf_kernel<<<cdiv(EL_VF, 256), 256, 0, stream>>>(hf_out, hf_bf);
    launch_gemm(4, VCDIM, VDIM, afc, VCDIM, 1, hf_bf,
                nullptr, 0, 0, nullptr, nullptr, nullptr, hc_ws, nullptr, N3S, stream);
    // 14) hs += N4 * relu(acs^T · hc)
    cvt_f32_bf16T_kernel<<<cdiv(EL_VCF, 256), 256, 0, stream>>>(hc_ws, hc_bf, VCDIM);
    launch_gemm(4, SDIM, VCDIM, acs, SDIM, 1, hc_bf,
                nullptr, 0, 0, nullptr, nullptr, nullptr, hs_ws, nullptr, N4S, stream);
    // 15) write hc / hs to d_out in [N,CO,node,L] order
    unpack_kernel<<<cdiv(EL_VCF, 256), 256, 0, stream>>>(hc_ws, hc_out, VCDIM);
    unpack_kernel<<<cdiv(EL_SF, 256), 256, 0, stream>>>(hs_ws, hs_out, SDIM);
}
